// ModKPCN_15109694947678
// MI455X (gfx1250) — matrix operand order repaired and checked by reading, not compiled, not run on hardware
//
#include <hip/hip_runtime.h>
#include <hip/hip_bf16.h>

typedef __attribute__((ext_vector_type(16))) __bf16 v16bf;
typedef __attribute__((ext_vector_type(8)))  float  v8f;

static constexpr int IMGD = 128;
static constexpr int NPIX = IMGD * IMGD;   // 16384

__device__ __forceinline__ unsigned short f2bf(float f) {
  unsigned u = __float_as_uint(f);
  u += 0x7FFFu + ((u >> 16) & 1u);         // round-to-nearest-even
  return (unsigned short)(u >> 16);
}
__device__ __forceinline__ float bf2f(unsigned short h) {
  return __uint_as_float(((unsigned)h) << 16);
}

// ---------------------------------------------------------------------------
// Pack OIHW f32 weights -> bf16, fragment-ordered for the 16-bit 16x32 A-matrix
// layout (ISA 7.12.2). Block = (tap t, ktile kt, mtile mt): 32 lanes x 16 bf16,
// lane-contiguous so each lane fetches its fragment with one 32B global load.
// A order used here (contiguous k, same as B): lane L: m = L%16, half = L/16; packed element e (0..15):
//   K = half*16 + e
// ---------------------------------------------------------------------------
__global__ void kpcn_pack_weights(const float* __restrict__ w,
                                  unsigned short* __restrict__ dst,
                                  int Cout, int Cin, int nKt, int nMt) {
  const long total = (long)9 * nKt * nMt * 512;
  for (long idx = blockIdx.x * (long)blockDim.x + threadIdx.x; idx < total;
       idx += (long)gridDim.x * blockDim.x) {
    int e    = (int)(idx & 15);
    int lane = (int)((idx >> 4) & 31);
    long blk = idx >> 9;
    int mt = (int)(blk % nMt);
    long r2 = blk / nMt;
    int kt = (int)(r2 % nKt);
    int t  = (int)(r2 / nKt);
    int hlf = lane >> 4, m = lane & 15;
    int v = e >> 1;
    int K = hlf * 16 + e;
    int ci = kt * 32 + K;
    int co = mt * 16 + m;
    float x = 0.0f;
    if (ci < Cin && co < Cout) x = w[((long)co * Cin + ci) * 9 + t];
    dst[idx] = f2bf(x);
  }
}

// ---------------------------------------------------------------------------
// Build bf16 network inputs in pixel-major [p][64] layout (ci contiguous):
// mode 0 = guidance (34 ch of kpcn_in); mode 1 = path stream (20 ch kpcn_in +
// mean over the 8 path samples). Channel index fastest -> coalesced stores.
// ---------------------------------------------------------------------------
__global__ void kpcn_prep_input(const float* __restrict__ kpcn,
                                const float* __restrict__ paths,
                                unsigned short* __restrict__ dst, int mode) {
  for (int idx = blockIdx.x * blockDim.x + threadIdx.x; idx < 64 * NPIX;
       idx += gridDim.x * blockDim.x) {
    int p  = idx >> 6;
    int ch = idx & 63;
    float v = 0.0f;
    if (mode == 0) {
      if (ch < 34) v = kpcn[ch * NPIX + p];
    } else {
      if (ch < 20) v = kpcn[ch * NPIX + p];
      else if (ch < 32) {
        int d = ch - 20;
        float s = 0.0f;
        #pragma unroll
        for (int q = 0; q < 8; ++q) s += paths[(q * 12 + d) * NPIX + p];
        v = s * 0.125f;
      }
    }
    dst[idx] = f2bf(v);
  }
}

// ---------------------------------------------------------------------------
// 3x3 same-pad conv as implicit GEMM on WMMA bf16.
// Activations live in global memory as [y][x][ci] (ci contiguous, CinPad a
// multiple of 32), so the LDS slab [row][col][ci] is staged with pure 16B
// vector copies (no transpose).
// Workgroup: 128 threads = 4 waves; block computes 64 out-channels x 64
// pixels. Wave w owns co-tile (4*blockIdx.y + w) and 4 accumulators (4 pixel
// groups of 16). Per (tap, ktile): 1 A-fragment global load (32B/lane) + 4
// B-fragment LDS loads + 4 independent WMMAs -> dense matrix-pipe usage.
// Epilogue: bias+ReLU, pack 8 consecutive co per lane, one b128 store.
// ---------------------------------------------------------------------------
__global__ __launch_bounds__(128)
void kpcn_conv3x3_wmma(const unsigned short* __restrict__ in, int CinPad,
                       const unsigned short* __restrict__ wpack, int nKt, int nMt,
                       const float* __restrict__ bias, int Cout, int CoutPad,
                       unsigned short* __restrict__ out, int relu) {
  __shared__ __align__(32) unsigned short lds[3 * 66 * 128];

  const int y     = blockIdx.x >> 1;          // image row
  const int cbase = (blockIdx.x & 1) << 6;    // 0 or 64

  // ---- stage 3 rows x 66 cols x CinPad, contiguous 16B chunks ----
  const int rowChunks = (66 * CinPad) >> 3;   // 8 ushorts per chunk
  for (int c = threadIdx.x; c < 3 * rowChunks; c += 128) {
    int r    = c / rowChunks;
    int cc   = c - r * rowChunks;
    int elem = cc << 3;
    int col  = elem / CinPad;
    int ci   = elem - col * CinPad;
    int yy = y + r - 1;
    int xx = cbase + col - 1;
    uint4 val = make_uint4(0u, 0u, 0u, 0u);
    if ((unsigned)yy < (unsigned)IMGD && (unsigned)xx < (unsigned)IMGD)
      val = *reinterpret_cast<const uint4*>(
          in + ((long)(yy * IMGD + xx)) * CinPad + ci);
    *reinterpret_cast<uint4*>(&lds[(r * 66) * CinPad + elem]) = val;
  }
  __syncthreads();

  const int lane = threadIdx.x & 31;
  const int wave = threadIdx.x >> 5;
  const int n    = lane & 15;
  const int hlf  = lane >> 4;
  const int mt   = (blockIdx.y << 2) + wave;  // this wave's 16-co tile

  v8f accs[4];
  #pragma unroll
  for (int g = 0; g < 4; ++g)
    accs[g] = (v8f){0.f, 0.f, 0.f, 0.f, 0.f, 0.f, 0.f, 0.f};

  for (int t = 0; t < 9; ++t) {
    const int ky = t / 3;
    const int kx = t - ky * 3;
    for (int kt = 0; kt < nKt; ++kt) {
      long wofs = ((long)((t * nKt + kt) * nMt + mt) * 32 + lane) * 16;
      v16bf a = *reinterpret_cast<const v16bf*>(wpack + wofs);
      const unsigned short* l0 =
          &lds[(ky * 66 + n + kx) * CinPad + kt * 32 + hlf * 16];
      v16bf b0 = *reinterpret_cast<const v16bf*>(l0);
      v16bf b1 = *reinterpret_cast<const v16bf*>(l0 + 16 * CinPad);
      v16bf b2 = *reinterpret_cast<const v16bf*>(l0 + 32 * CinPad);
      v16bf b3 = *reinterpret_cast<const v16bf*>(l0 + 48 * CinPad);
      accs[0] = __builtin_amdgcn_wmma_f32_16x16x32_bf16(
          false, a, false, b0, (short)0, accs[0], false, false);
      accs[1] = __builtin_amdgcn_wmma_f32_16x16x32_bf16(
          false, a, false, b1, (short)0, accs[1], false, false);
      accs[2] = __builtin_amdgcn_wmma_f32_16x16x32_bf16(
          false, a, false, b2, (short)0, accs[2], false, false);
      accs[3] = __builtin_amdgcn_wmma_f32_16x16x32_bf16(
          false, a, false, b3, (short)0, accs[3], false, false);
    }
  }

  // ---- epilogue: C/D layout -> lane holds co = 16*mt + 8*half + v, 8 wide ----
  const int cob = (mt << 4) + (hlf << 3);
  #pragma unroll
  for (int g = 0; g < 4; ++g) {
    int px = cbase + (g << 4) + n;
    unsigned d[4];
    #pragma unroll
    for (int dw = 0; dw < 4; ++dw) {
      unsigned lo, hi;
      {
        int co = cob + 2 * dw;
        float val = 0.0f;
        if (co < Cout) {
          val = accs[g][2 * dw] + bias[co];
          if (relu) val = fmaxf(val, 0.0f);
        }
        lo = f2bf(val);
      }
      {
        int co = cob + 2 * dw + 1;
        float val = 0.0f;
        if (co < Cout) {
          val = accs[g][2 * dw + 1] + bias[co];
          if (relu) val = fmaxf(val, 0.0f);
        }
        hi = f2bf(val);
      }
      d[dw] = lo | (hi << 16);
    }
    *reinterpret_cast<uint4*>(out + ((long)(y * IMGD + px)) * CoutPad + cob) =
        make_uint4(d[0], d[1], d[2], d[3]);
  }
}

// ---------------------------------------------------------------------------
// Per-pixel online softmax stats over 441 bf16 logits ([p][448] layout).
// ---------------------------------------------------------------------------
__global__ void kpcn_softmax_stats(const unsigned short* __restrict__ logits,
                                   float* __restrict__ Mx, float* __restrict__ Sx) {
  int p = blockIdx.x * blockDim.x + threadIdx.x;
  if (p >= NPIX) return;
  const unsigned short* row = logits + (long)p * 448;
  float m = -3.0e38f, s = 0.0f;
  for (int k = 0; k < 441; ++k) {
    float x  = bf2f(row[k]);
    float nm = fmaxf(m, x);
    s = s * __expf(m - nm) + __expf(x - nm);
    m = nm;
  }
  Mx[p] = m;
  Sx[p] = s;
}

// ---------------------------------------------------------------------------
// Splat as gather: out[y,x] = sum_k wnorm[k, y-dy, x-dx] * data[y-dy, x-dx].
// ---------------------------------------------------------------------------
__global__ void kpcn_splat(const unsigned short* __restrict__ logits,
                           const float* __restrict__ Mx, const float* __restrict__ Sx,
                           const float* __restrict__ data, float* __restrict__ outp) {
  int p = blockIdx.x * blockDim.x + threadIdx.x;
  if (p >= NPIX) return;
  int y = p >> 7, x = p & 127;
  float a0 = 0.f, a1 = 0.f, a2 = 0.f;
  for (int k = 0; k < 441; ++k) {
    int ky = k / 21;
    int dy = ky - 10;
    int dx = (k - ky * 21) - 10;
    int ys = y - dy, xs = x - dx;
    if ((unsigned)ys < 128u && (unsigned)xs < 128u) {
      int sp  = (ys << 7) + xs;
      float w = __expf(bf2f(logits[(long)sp * 448 + k]) - Mx[sp]) / Sx[sp];
      a0 = fmaf(w, data[sp], a0);
      a1 = fmaf(w, data[NPIX + sp], a1);
      a2 = fmaf(w, data[2 * NPIX + sp], a2);
    }
  }
  outp[p] = a0;
  outp[NPIX + p] = a1;
  outp[2 * NPIX + p] = a2;
}

// ---------------------------------------------------------------------------
extern "C" void kernel_launch(void* const* d_in, const int* in_sizes, int n_in,
                              void* d_out, int out_size, void* d_ws, size_t ws_size,
                              hipStream_t stream) {
  const float* kpcn  = (const float*)d_in[0];
  const float* paths = (const float*)d_in[1];
  const float* rad   = (const float*)d_in[2];
  const float* W0[2] = {(const float*)d_in[3], (const float*)d_in[9]};
  const float* B0[2] = {(const float*)d_in[4], (const float*)d_in[10]};
  const float* WM[2] = {(const float*)d_in[5], (const float*)d_in[11]};
  const float* BM[2] = {(const float*)d_in[6], (const float*)d_in[12]};
  const float* WL[2] = {(const float*)d_in[7], (const float*)d_in[13]};
  const float* BL[2] = {(const float*)d_in[8], (const float*)d_in[14]};
  const int CIN0[2]  = {34, 32};

  // ---- workspace carve (all 256B aligned; ~35 MB total, L2-resident) ----
  char* ws = (char*)d_ws;
  size_t off = 0;
  auto carve = [&](size_t bytes) -> void* {
    void* p = ws + off;
    off += (bytes + 255) & ~(size_t)255;
    return p;
  };
  unsigned short *wp0[2], *wpm[2], *wpl[2];
  for (int b = 0; b < 2; ++b) {
    wp0[b] = (unsigned short*)carve(9ull * 2 * 8 * 512 * 2);        // 147456 B
    wpm[b] = (unsigned short*)carve(16ull * 9 * 4 * 8 * 512 * 2);   // 4.7 MB
    wpl[b] = (unsigned short*)carve(9ull * 4 * 28 * 512 * 2);       // 1.03 MB
  }
  unsigned short* inb[2];
  inb[0] = (unsigned short*)carve(64ull * NPIX * 2);
  inb[1] = (unsigned short*)carve(64ull * NPIX * 2);
  unsigned short* actA = (unsigned short*)carve(448ull * NPIX * 2); // logits-capable
  unsigned short* actB = (unsigned short*)carve(128ull * NPIX * 2);
  float* Mx = (float*)carve(NPIX * 4);
  float* Sx = (float*)carve(NPIX * 4);

  // ---- weight packing + input prep ----
  for (int b = 0; b < 2; ++b) {
    kpcn_pack_weights<<<288, 256, 0, stream>>>(W0[b], wp0[b], 100, CIN0[b], 2, 8);
    for (int l = 0; l < 16; ++l)
      kpcn_pack_weights<<<576, 256, 0, stream>>>(
          WM[b] + (long)l * 100 * 100 * 9, wpm[b] + (long)l * 147456, 100, 100, 4, 8);
    kpcn_pack_weights<<<2016, 256, 0, stream>>>(WL[b], wpl[b], 441, 100, 4, 28);
    kpcn_prep_input<<<4096, 256, 0, stream>>>(kpcn, paths, inb[b], b);
  }

  // ---- conv chains + softmax splat ----
  float* outF = (float*)d_out;
  for (int b = 0; b < 2; ++b) {
    // layer 0: CinPad 64 -> Cout 100 (pad 128); grid.y = 128/64 = 2
    kpcn_conv3x3_wmma<<<dim3(256, 2), 128, 0, stream>>>(
        inb[b], 64, wp0[b], 2, 8, B0[b], 100, 128, actB, 1);
    const unsigned short* cur = actB;
    for (int l = 0; l < 16; ++l) {
      unsigned short* nxt = (l & 1) ? actB : actA;
      kpcn_conv3x3_wmma<<<dim3(256, 2), 128, 0, stream>>>(
          cur, 128, wpm[b] + (long)l * 147456, 4, 8, BM[b] + l * 100, 100, 128,
          nxt, 1);
      cur = nxt;
    }
    // last layer: 100(pad 128) -> 441(pad 448); grid.y = 448/64 = 7; no relu
    kpcn_conv3x3_wmma<<<dim3(256, 7), 128, 0, stream>>>(
        cur, 128, wpl[b], 4, 28, BL[b], 441, 448, actA, 0);
    kpcn_softmax_stats<<<64, 256, 0, stream>>>(actA, Mx, Sx);
    kpcn_splat<<<64, 256, 0, stream>>>(actA, Mx, Sx, rad, outF + (long)b * 3 * NPIX);
  }
}
